// NSAModel_52527450030127
// MI455X (gfx1250) — matrix operand order repaired and checked by reading, not compiled, not run on hardware
//
#include <hip/hip_runtime.h>
#include <hip/hip_bf16.h>
#include <math.h>

// ---- model dims (fixed by reference) ----
#define MTOK 2048      // B*S
#define HDIM 768
#define DFFD 3072
#define VOC  21128
#define VOCPAD 21184   // VOC rounded up to 64
#define NLAYER 6
#define NEXP 4
#define CAPTOK 640     // int(B*S/NE * 1.25)

typedef __bf16 bf16;
typedef __attribute__((ext_vector_type(16))) __bf16 v16bf;
typedef __attribute__((ext_vector_type(8)))  __bf16 v8bf;
typedef __attribute__((ext_vector_type(8)))  float  v8f;

__device__ __forceinline__ float gelu_f(float x) {
  // exact erf GELU (reference uses approximate=False)
  return 0.5f * x * (1.0f + erff(x * 0.70710678118654752440f));
}
__device__ __forceinline__ float clip100(float x) {
  return fminf(100.0f, fmaxf(-100.0f, x));
}

// ======================================================================
// WMMA GEMM: C[M,N] (f32) = A[M,K] (bf16, row-major) x W (via Wt[Npad,K])
// Wt is zero-padded to Npad (multiple of 64) rows, so the K-loop has NO
// guards: pure loads + 8x v_wmma per 32-wide K-step (32x64 tile per wave,
// 2 A fragments reuse each of the 4 B fragments).
// EPI: 0 = +bias ; 1 = +bias,GELU ; 2 = +bias, *1/sqrt(H), clip +-100
// ======================================================================
template <int EPI>
__global__ __launch_bounds__(256) void gemm_wmma_bf16(
    const bf16* __restrict__ A, const bf16* __restrict__ Wt,
    const float* __restrict__ bias, float* __restrict__ C,
    int M, int N, int Npad, int K)
{
  const int lane = threadIdx.x & 31;
  const int half = lane >> 4;        // 0: lanes 0-15, 1: lanes 16-31
  const int l16  = lane & 15;
  const int wid  = blockIdx.x * 8 + (threadIdx.x >> 5);
  const int mTiles  = M >> 5;        // 32-row tiles
  const int nChunks = Npad >> 6;     // 64-col chunks
  if (wid >= mTiles * nChunks) return;
  const int m0 = (wid % mTiles) << 5;
  const int n0 = (wid / mTiles) << 6;

  // A fragment in contiguous K order (matches B):
  // lanes 0-15: K = kb+0..15; lanes 16-31: K = kb+16..31.
  const bf16* aRow0 = A + (size_t)(m0 + l16) * K + (half ? 16 : 0);
  const bf16* aRow1 = aRow0 + (size_t)16 * K;
  // B fragment in the same contiguous K order as A: lanes 0-15 hold K=kb..kb+15,
  // lanes 16-31 hold K=kb+16..kb+31 -> contiguous 32B read from Wt[Npad,K].
  const bf16* wRow = Wt + (size_t)(n0 + l16) * K + (half ? 16 : 0);

  v8f acc[8];
  #pragma unroll
  for (int j = 0; j < 8; ++j)
    #pragma unroll
    for (int r = 0; r < 8; ++r) acc[j][r] = 0.0f;

  for (int kb = 0; kb < K; kb += 32) {
    union { v16bf v; v8bf h[2]; } a0, a1;
    a0.h[0] = *(const v8bf*)(aRow0 + kb);
    a0.h[1] = *(const v8bf*)(aRow0 + kb + 8);
    a1.h[0] = *(const v8bf*)(aRow1 + kb);
    a1.h[1] = *(const v8bf*)(aRow1 + kb + 8);
    #pragma unroll
    for (int j = 0; j < 4; ++j) {
      const v16bf bfrag = *(const v16bf*)(wRow + (size_t)j * 16 * K + kb);
      acc[j] = __builtin_amdgcn_wmma_f32_16x16x32_bf16(
          false, a0.v, false, bfrag, (short)0, acc[j], false, false);
      acc[4 + j] = __builtin_amdgcn_wmma_f32_16x16x32_bf16(
          false, a1.v, false, bfrag, (short)0, acc[4 + j], false, false);
    }
  }

  // C/D layout: VGPR r -> row +r+8*half ; col n0 + 16j + l16
  #pragma unroll
  for (int j = 0; j < 4; ++j) {
    const int n = n0 + j * 16 + l16;
    if (n >= N) continue;
    const float bv = bias ? bias[n] : 0.0f;
    #pragma unroll
    for (int part = 0; part < 2; ++part) {
      #pragma unroll
      for (int r = 0; r < 8; ++r) {
        float v = acc[part * 4 + j][r] + bv;
        if (EPI == 1)      v = gelu_f(v);
        else if (EPI == 2) v = clip100(v * 0.03608439182435161f); // 1/sqrt(768)
        C[(size_t)(m0 + part * 16 + r + half * 8) * N + n] = v;
      }
    }
  }
}

// ======================================================================
// Support kernels
// ======================================================================
__global__ void f32_to_bf16_kernel(const float* __restrict__ in,
                                   bf16* __restrict__ out, size_t n) {
  size_t i = (size_t)blockIdx.x * 256 + threadIdx.x;
  if (i < n) out[i] = (bf16)in[i];
}

// W[K,N] f32 (din,dout layout) -> Wt[Npad,K] bf16, zero-padded rows N..Npad
__global__ void wt_transpose_kernel(const float* __restrict__ W,
                                    bf16* __restrict__ Wt, int K, int N,
                                    int Npad) {
  size_t i = (size_t)blockIdx.x * 256 + threadIdx.x;
  if (i >= (size_t)Npad * K) return;
  int n = (int)(i / K);
  int k = (int)(i % K);
  Wt[i] = (n < N) ? (bf16)W[(size_t)k * N + n] : (bf16)0.0f;
}

__global__ void embed_kernel(const int* __restrict__ ids,
                             const float* __restrict__ emb,
                             const float* __restrict__ pos,
                             float* __restrict__ out) {
  const int t = blockIdx.x;            // token index, row-major [B,S]
  const int id = ids[t];
  const int s = t & 511;               // S = 512
  for (int c = threadIdx.x; c < HDIM; c += 256) {
    float e = clip100(emb[(size_t)id * HDIM + c]);
    float p = clip100(pos[(size_t)s * HDIM + c]);
    out[(size_t)t * HDIM + c] = e + p;
  }
}

// rowwise LayerNorm of (ca*ina + cb*inb), optional affine
__global__ __launch_bounds__(256) void ln_kernel(
    const float* __restrict__ ina, const float* __restrict__ inb,
    float ca, float cb, const float* __restrict__ g,
    const float* __restrict__ b, float eps, float* __restrict__ out,
    int ncols) {
  __shared__ float r1[256], r2[256];
  const size_t base = (size_t)blockIdx.x * ncols;
  float s1 = 0.0f, s2 = 0.0f;
  for (int c = threadIdx.x; c < ncols; c += 256) {
    float v = ca * ina[base + c];
    if (inb) v += cb * inb[base + c];
    s1 += v; s2 += v * v;
  }
  r1[threadIdx.x] = s1; r2[threadIdx.x] = s2;
  __syncthreads();
  for (int off = 128; off > 0; off >>= 1) {
    if ((int)threadIdx.x < off) {
      r1[threadIdx.x] += r1[threadIdx.x + off];
      r2[threadIdx.x] += r2[threadIdx.x + off];
    }
    __syncthreads();
  }
  const float mu  = r1[0] / ncols;
  const float var = fmaxf(r2[0] / ncols - mu * mu, 0.0f);
  const float inv = rsqrtf(var + eps);
  for (int c = threadIdx.x; c < ncols; c += 256) {
    float v = ca * ina[base + c];
    if (inb) v += cb * inb[base + c];
    float y = (v - mu) * inv;
    if (g) y = y * g[c] + b[c];
    out[base + c] = y;
  }
}

// logits[t,e] = dot(g1[t,:], rw2[:,e]) + rb2[e]   (N=4, VALU is fine)
__global__ void router_logits_kernel(const float* __restrict__ g1,
                                     const float* __restrict__ rw2,
                                     const float* __restrict__ rb2,
                                     float* __restrict__ lg) {
  int i = blockIdx.x * 256 + threadIdx.x;
  if (i >= MTOK * NEXP) return;
  const int t = i >> 2, e = i & 3;
  float s = rb2[e];
  const float* row = g1 + (size_t)t * HDIM;
  for (int k = 0; k < HDIM; ++k) s += row[k] * rw2[k * NEXP + e];
  lg[i] = s;
}

// top-2 + softmax + FCFS capacity drop; serial cumsum over tokens (tiny)
__global__ void router_select_kernel(const float* __restrict__ lg,
                                     float* __restrict__ wsum) {
  if (blockIdx.x != 0 || threadIdx.x != 0) return;
  int c0[NEXP] = {0, 0, 0, 0}, c1[NEXP] = {0, 0, 0, 0};
  for (int t = 0; t < MTOK; ++t) {
    const float* L = lg + t * NEXP;
    float* W = wsum + t * NEXP;
    for (int e = 0; e < NEXP; ++e) W[e] = 0.0f;
    int i0 = 0; float v0 = L[0];
    for (int e = 1; e < NEXP; ++e) if (L[e] > v0) { v0 = L[e]; i0 = e; }
    int i1 = -1; float v1 = -3.4e38f;
    for (int e = 0; e < NEXP; ++e)
      if (e != i0 && L[e] > v1) { v1 = L[e]; i1 = e; }
    const float ex = expf(v1 - v0);
    const float den = 1.0f / (1.0f + ex);
    if (++c0[i0] <= CAPTOK) W[i0] += den;        // softmax weight of top-1
    if (++c1[i1] <= CAPTOK) W[i1] += ex * den;   // softmax weight of top-2
  }
}

__global__ void zero_kernel(float* p, size_t n) {
  size_t i = (size_t)blockIdx.x * 256 + threadIdx.x;
  if (i < n) p[i] = 0.0f;
}
__global__ void accum_routed_kernel(float* __restrict__ acc,
                                    const float* __restrict__ x,
                                    const float* __restrict__ wsum, int j,
                                    size_t n) {
  size_t i = (size_t)blockIdx.x * 256 + threadIdx.x;
  if (i >= n) return;
  size_t t = i / HDIM;
  acc[i] += wsum[t * NEXP + j] * x[i];
}
__global__ void accum_scalar_kernel(float* __restrict__ acc,
                                    const float* __restrict__ x, float c,
                                    size_t n) {
  size_t i = (size_t)blockIdx.x * 256 + threadIdx.x;
  if (i < n) acc[i] += c * x[i];
}
__global__ void mix_clip_kernel(float* __restrict__ h,
                                const float* __restrict__ blk, size_t n) {
  size_t i = (size_t)blockIdx.x * 256 + threadIdx.x;
  if (i < n) h[i] = clip100(0.5f * h[i] + 0.5f * blk[i]);
}

// ======================================================================
// Orchestration
// Input flattening assumed jax-pytree style (sorted dict keys, lists in
// order): 0=input_ids, 1=emb, 2..229 = 6 layers x 38 leaves,
// 230=lm_b, 231=lm_w, 232=ln_b, 233=ln_g, 234=pos.
// Per-layer leaves: fb,fw,ib,iw,ln1_b,ln1_g,ln2_b,ln2_g,ob,ow,rb1,rb2,
// routed[0..3]{b1,b2,w1,w2}, rw1, rw2, shared[0..1]{b1,b2,w1,w2}.
// ======================================================================
extern "C" void kernel_launch(void* const* d_in, const int* in_sizes, int n_in,
                              void* d_out, int out_size, void* d_ws,
                              size_t ws_size, hipStream_t stream) {
  (void)in_sizes; (void)n_in; (void)out_size; (void)ws_size;
  auto F = [&](int i) { return (const float*)d_in[i]; };
  const int* ids = (const int*)d_in[0];
  const float* emb = F(1);
  const int LSTRIDE = 38;
  const int TAIL = 2 + NLAYER * LSTRIDE;
  const float* lm_b = F(TAIL + 0);
  const float* lm_w = F(TAIL + 1);
  const float* ln_b = F(TAIL + 2);
  const float* ln_g = F(TAIL + 3);
  const float* pos  = F(TAIL + 4);

  // workspace carve (256B aligned)
  char* p = (char*)d_ws;
  auto carve = [&](size_t bytes) {
    char* r = p; p += (bytes + 255) & ~(size_t)255; return (void*)r;
  };
  float* h   = (float*)carve((size_t)MTOK * HDIM * 4);
  float* a   = (float*)carve((size_t)MTOK * HDIM * 4);
  float* h1  = (float*)carve((size_t)MTOK * HDIM * 4);
  float* t1  = (float*)carve((size_t)MTOK * DFFD * 4);
  float* t2  = (float*)carve((size_t)MTOK * HDIM * 4);
  float* t3  = (float*)carve((size_t)MTOK * HDIM * 4);
  bf16* xbh  = (bf16*)carve((size_t)MTOK * HDIM * 2);
  bf16* xbd  = (bf16*)carve((size_t)MTOK * DFFD * 2);
  bf16* wb   = (bf16*)carve((size_t)VOCPAD * HDIM * 2);
  float* lg   = (float*)carve((size_t)MTOK * NEXP * 4);
  float* wsum = (float*)carve((size_t)MTOK * NEXP * 4);

  auto eb = [](size_t n) { return (int)((n + 255) / 256); };
  auto to_bf16 = [&](const float* in, bf16* o, size_t n) {
    f32_to_bf16_kernel<<<eb(n), 256, 0, stream>>>(in, o, n);
  };
  auto conv_wt = [&](const float* W, int K, int N) {
    int Npad = (N + 63) & ~63;
    size_t n = (size_t)Npad * K;
    wt_transpose_kernel<<<eb(n), 256, 0, stream>>>(W, wb, K, N, Npad);
  };
  auto gemm = [&](const bf16* A, const float* bias, float* C, int M, int N,
                  int K, int epi) {
    int Npad = (N + 63) & ~63;
    int waves = (M >> 5) * (Npad >> 6);
    int blocks = (waves + 7) / 8;
    if (epi == 0)
      gemm_wmma_bf16<0><<<blocks, 256, 0, stream>>>(A, wb, bias, C, M, N,
                                                    Npad, K);
    else if (epi == 1)
      gemm_wmma_bf16<1><<<blocks, 256, 0, stream>>>(A, wb, bias, C, M, N,
                                                    Npad, K);
    else
      gemm_wmma_bf16<2><<<blocks, 256, 0, stream>>>(A, wb, bias, C, M, N,
                                                    Npad, K);
  };
  const size_t NH = (size_t)MTOK * HDIM;
  const size_t ND = (size_t)MTOK * DFFD;

  // embedding + input LN
  embed_kernel<<<MTOK, 256, 0, stream>>>(ids, emb, pos, t1);
  ln_kernel<<<MTOK, 256, 0, stream>>>(t1, nullptr, 1.f, 0.f, ln_g, ln_b,
                                      1e-5f, h, HDIM);

  for (int l = 0; l < NLAYER; ++l) {
    const int B0 = 2 + l * LSTRIDE;
    const float* fb = F(B0 + 0);  const float* fw = F(B0 + 1);
    const float* ib = F(B0 + 2);  const float* iw = F(B0 + 3);
    const float* ln1b = F(B0 + 4); const float* ln1g = F(B0 + 5);
    const float* ln2b = F(B0 + 6); const float* ln2g = F(B0 + 7);
    const float* ob = F(B0 + 8);  const float* ow = F(B0 + 9);
    const float* rb1 = F(B0 + 10); const float* rb2 = F(B0 + 11);
    const float* rw1 = F(B0 + 28); const float* rw2 = F(B0 + 29);

    to_bf16(h, xbh, NH);

    // router: t1 = gelu(x@rw1+rb1); logits = t1@rw2+rb2; select top-2/cap
    conv_wt(rw1, HDIM, HDIM);
    gemm(xbh, rb1, t1, MTOK, HDIM, HDIM, 1);
    router_logits_kernel<<<eb(MTOK * NEXP), 256, 0, stream>>>(t1, rw2, rb2, lg);
    router_select_kernel<<<1, 32, 0, stream>>>(lg, wsum);

    zero_kernel<<<eb(NH), 256, 0, stream>>>(t2, NH);

    for (int j = 0; j < NEXP; ++j) {
      const float* b1 = F(B0 + 12 + j * 4 + 0);
      const float* b2 = F(B0 + 12 + j * 4 + 1);
      const float* w1 = F(B0 + 12 + j * 4 + 2);
      const float* w2 = F(B0 + 12 + j * 4 + 3);
      conv_wt(w1, HDIM, DFFD);
      gemm(xbh, b1, t1, MTOK, DFFD, HDIM, 1);
      to_bf16(t1, xbd, ND);
      conv_wt(w2, DFFD, HDIM);
      gemm(xbd, b2, t3, MTOK, HDIM, DFFD, 0);
      accum_routed_kernel<<<eb(NH), 256, 0, stream>>>(t2, t3, wsum, j, NH);
    }
    for (int s = 0; s < 2; ++s) {
      const float* b1 = F(B0 + 30 + s * 4 + 0);
      const float* b2 = F(B0 + 30 + s * 4 + 1);
      const float* w1 = F(B0 + 30 + s * 4 + 2);
      const float* w2 = F(B0 + 30 + s * 4 + 3);
      conv_wt(w1, HDIM, DFFD);
      gemm(xbh, b1, t1, MTOK, DFFD, HDIM, 1);
      to_bf16(t1, xbd, ND);
      conv_wt(w2, DFFD, HDIM);
      gemm(xbd, b2, t3, MTOK, HDIM, DFFD, 0);
      accum_scalar_kernel<<<eb(NH), 256, 0, stream>>>(t2, t3, 0.5f, NH);
    }

    // output proj, residual mix, LN (no affine, eps 1e-6)
    to_bf16(t2, xbh, NH);
    conv_wt(ow, HDIM, HDIM);
    gemm(xbh, ob, t3, MTOK, HDIM, HDIM, 0);
    ln_kernel<<<MTOK, 256, 0, stream>>>(t3, h, 0.5f, 0.5f, nullptr, nullptr,
                                        1e-6f, a, HDIM);
    // h1 = LN(h + a, ln1, 1e-5)
    ln_kernel<<<MTOK, 256, 0, stream>>>(h, a, 1.f, 1.f, ln1g, ln1b, 1e-5f, h1,
                                        HDIM);
    // FFN
    to_bf16(h1, xbh, NH);
    conv_wt(iw, HDIM, DFFD);
    gemm(xbh, ib, t1, MTOK, DFFD, HDIM, 1);
    to_bf16(t1, xbd, ND);
    conv_wt(fw, DFFD, HDIM);
    gemm(xbd, fb, t3, MTOK, HDIM, DFFD, 0);
    // block out = LN(h1 + ff, ln2, 1e-5); h = clip(0.5h + 0.5 blk)
    ln_kernel<<<MTOK, 256, 0, stream>>>(h1, t3, 1.f, 1.f, ln2g, ln2b, 1e-5f,
                                        t2, HDIM);
    mix_clip_kernel<<<eb(NH), 256, 0, stream>>>(h, t2, NH);
  }

  // LM head: logits = clip((h@lm_w + lm_b) / sqrt(H))
  to_bf16(h, xbh, NH);
  conv_wt(lm_w, HDIM, VOC);
  gemm(xbh, lm_b, (float*)d_out, MTOK, VOC, HDIM, 2);
}
